// EGNN_NEC_50654844289863
// MI455X (gfx1250) — matrix operand order repaired and checked by reading, not compiled, not run on hardware
//
#include <hip/hip_runtime.h>

// ---------------------------------------------------------------------------
// EGNN forward for MI455X (gfx1250, wave32, WMMA 16x16x32 f16 -> f32 accum).
// Memory-bound (~1 GB traffic @ 23.3 TB/s vs 45 GFLOP): h and GEMM operands
// kept in f16, edge MLP fully fused, segment_sum via global f32 atomics.
// Weights staged to LDS via async global->LDS loads (ASYNCcnt) when available.
// ---------------------------------------------------------------------------

#define DEVFN __device__ __forceinline__

typedef _Float16 half16 __attribute__((ext_vector_type(16)));
typedef _Float16 half8  __attribute__((ext_vector_type(8)));
typedef float    float8 __attribute__((ext_vector_type(8)));
typedef int      int4v  __attribute__((ext_vector_type(4)));

static constexpr int NN = 100000;
static constexpr int EE = 1200000;

// workspace layout (bytes, all 16B-aligned)
static constexpr size_t OFF_H    = 0;                        // N*64 half
static constexpr size_t OFF_AGGM = (size_t)NN * 64 * 2;      // N*64 f32
static constexpr size_t OFF_AGGX = OFF_AGGM + (size_t)NN * 64 * 4;  // N*3 f32
static constexpr size_t OFF_CNT  = OFF_AGGX + (size_t)NN * 3 * 4;   // N f32
static constexpr size_t OFF_WE1T = OFF_CNT + (size_t)NN * 4;        // 64x160 half
static constexpr size_t OFF_WE2T = OFF_WE1T + 64 * 160 * 2;         // 64x64 half
static constexpr size_t OFF_WC1T = OFF_WE2T + 64 * 64 * 2;
static constexpr size_t OFF_WN1T = OFF_WC1T + 64 * 64 * 2;          // 64x128 half
static constexpr size_t OFF_WN2T = OFF_WN1T + 64 * 128 * 2;
static constexpr size_t OFF_WOT  = OFF_WN2T + 64 * 64 * 2;
static constexpr size_t OFF_WIT  = OFF_WOT + 64 * 64 * 2;           // 64x32 half

#if __has_builtin(__builtin_amdgcn_global_load_async_to_lds_b128) && \
    __has_builtin(__builtin_amdgcn_s_wait_asynccnt)
#define HAVE_ASYNC_LDS 1
#endif

DEVFN float8 wmma_f16(half16 a, half16 b, float8 c) {
    return __builtin_amdgcn_wmma_f32_16x16x32_f16(false, a, false, b, (short)0,
                                                  c, false, false);
}

DEVFN half16 packA(half8 lo, half8 hi) {
    union { half16 v; half8 h[2]; } u;
    u.h[0] = lo; u.h[1] = hi;
    return u.v;
}

DEVFN half8 ld8h(const _Float16* p) { return *(const half8*)p; }

DEVFN half8 cvt8(const float* p) {
    float4 a = *(const float4*)p;
    float4 b = *(const float4*)(p + 4);
    half8 r;
    r[0] = (_Float16)a.x; r[1] = (_Float16)a.y; r[2] = (_Float16)a.z; r[3] = (_Float16)a.w;
    r[4] = (_Float16)b.x; r[5] = (_Float16)b.y; r[6] = (_Float16)b.z; r[7] = (_Float16)b.w;
    return r;
}

// B fragment for 16x16x32 WMMA from transposed half weights Wt[Ncol][Kpad]:
// lane&15 = output column within tile nt, lane>>4 selects K octets (8kh..8kh+7, 16+8kh..16+8kh+7).
DEVFN half16 ldsB(const _Float16* W, int Kpad, int nt, int s, int lane) {
    int n  = nt * 16 + (lane & 15);
    int kh = lane >> 4;
    const _Float16* q = W + n * Kpad + s * 32;
    return packA(ld8h(q + 8 * kh), ld8h(q + 16 + 8 * kh));
}

DEVFN void atomAddF(float* p, float v) {
    (void)__hip_atomic_fetch_add(p, v, __ATOMIC_RELAXED, __HIP_MEMORY_SCOPE_AGENT);
}

// Stage weights global -> LDS.  Async (no VGPR round-trip, ASYNCcnt) when the
// toolchain exposes the gfx1250 builtin; plain copy otherwise.
DEVFN void stage_lds(_Float16* dst, const _Float16* src, int nhalf, int tid, int nth) {
#ifdef HAVE_ASYNC_LDS
    int n = nhalf >> 3;  // 16B chunks
    for (int i = tid; i < n; i += nth) {
        __builtin_amdgcn_global_load_async_to_lds_b128(
            (__attribute__((address_space(1))) int4v*)(src + i * 8),
            (__attribute__((address_space(3))) int4v*)(dst + i * 8), 0, 0);
    }
#else
    unsigned int* d = (unsigned int*)dst;
    const unsigned int* s = (const unsigned int*)src;
    int n = nhalf >> 1;
    for (int i = tid; i < n; i += nth) d[i] = s[i];
#endif
}

DEVFN void stage_fence() {
#ifdef HAVE_ASYNC_LDS
    __builtin_amdgcn_s_wait_asynccnt(0);
#endif
    __syncthreads();
}

// ---------------------------------------------------------------------------
// prep: W[K][Ncol] f32 -> Wt[Ncol][Kpad] half, zero padded over K
// ---------------------------------------------------------------------------
__global__ void transpose_to_half(const float* __restrict__ src, _Float16* __restrict__ dst,
                                  int K, int Ncol, int Kpad) {
    int i = blockIdx.x * blockDim.x + threadIdx.x;
    if (i >= Ncol * Kpad) return;
    int n = i / Kpad, k = i % Kpad;
    dst[i] = (k < K) ? (_Float16)src[k * Ncol + n] : (_Float16)0.0f;
}

// ---------------------------------------------------------------------------
// h = node_feats @ W_in + b_in   (stored f16)
// ---------------------------------------------------------------------------
__global__ void __launch_bounds__(256)
egnn_embed(const float* __restrict__ nf, const _Float16* __restrict__ Wint,
           const float* __restrict__ b_in, _Float16* __restrict__ hbuf, int Ntot) {
    __shared__ __align__(32) _Float16 sW[64 * 32];
    stage_lds(sW, Wint, 64 * 32, threadIdx.x, blockDim.x);
    stage_fence();

    int wv = threadIdx.x >> 5, ln = threadIdx.x & 31;
    int n0 = (blockIdx.x * 8 + wv) * 16;
    if (n0 >= Ntot) return;
    int kh = ln >> 4, sub = ln & 15;
    int node = n0 + sub;

    const float* p = nf + (size_t)node * 32;
    half16 A = packA(cvt8(p + 8 * kh), cvt8(p + 16 + 8 * kh));

    float8 acc[4];
#pragma unroll
    for (int nt = 0; nt < 4; ++nt) {
        float8 c = {};
        acc[nt] = wmma_f16(A, ldsB(sW, 32, nt, 0, ln), c);
    }
#pragma unroll
    for (int nt = 0; nt < 4; ++nt) {
        int col = nt * 16 + sub;
        float b = b_in[col];
#pragma unroll
        for (int v = 0; v < 8; ++v)
            hbuf[(size_t)(n0 + v + 8 * kh) * 64 + col] = (_Float16)(acc[nt][v] + b);
    }
}

// ---------------------------------------------------------------------------
// fused edge kernel: edge MLP (K=160 -> 64 -> 64), coord MLP (64->64->1),
// atomic segment sums for agg_m, agg_x, cnt.  One wave = 16 edges, 8 waves/WG.
// ---------------------------------------------------------------------------
__global__ void __launch_bounds__(256)
egnn_edge(const _Float16* __restrict__ hbuf, const float* __restrict__ edge_attr,
          const float* __restrict__ coords, const int* __restrict__ edge_index,
          const _Float16* __restrict__ We1t, const float* __restrict__ be1,
          const _Float16* __restrict__ We2t, const float* __restrict__ be2,
          const _Float16* __restrict__ Wc1t, const float* __restrict__ bc1,
          const float* __restrict__ Wc2, const float* __restrict__ bc2,
          float* __restrict__ aggm, float* __restrict__ aggx,
          float* __restrict__ cntbuf, int Etot) {
    __shared__ __align__(32) _Float16 sWe1[64 * 160];
    __shared__ __align__(32) _Float16 sWe2[64 * 64];
    __shared__ __align__(32) _Float16 sWc1[64 * 64];
    __shared__ __align__(32) _Float16 sY[8][16 * 64];
    __shared__ float sDr[8][16][3];
    __shared__ int   sRow[8][16];

    stage_lds(sWe1, We1t, 64 * 160, threadIdx.x, blockDim.x);
    stage_lds(sWe2, We2t, 64 * 64, threadIdx.x, blockDim.x);
    stage_lds(sWc1, Wc1t, 64 * 64, threadIdx.x, blockDim.x);
    stage_fence();

    int wv = threadIdx.x >> 5, ln = threadIdx.x & 31;
    int kh = ln >> 4, sub = ln & 15;
    int e0 = (blockIdx.x * 8 + wv) * 16;
    if (e0 >= Etot) return;
    int e = e0 + sub;

    int r = edge_index[e];
    int c = edge_index[Etot + e];

    float dx = coords[r * 3 + 0] - coords[c * 3 + 0];
    float dy = coords[r * 3 + 1] - coords[c * 3 + 1];
    float dz = coords[r * 3 + 2] - coords[c * 3 + 2];
    float radial = dx * dx + dy * dy + dz * dz;
    if (kh == 0) {
        sDr[wv][sub][0] = dx; sDr[wv][sub][1] = dy; sDr[wv][sub][2] = dz;
        sRow[wv][sub] = r;
    }

    float ea[16];
    const float* eap = edge_attr + (size_t)e * 16;
#pragma unroll
    for (int i = 0; i < 16; ++i) ea[i] = eap[i];

    // ---- GEMM1: x[16x160] @ We1[160x64]  (x = [h_r | h_c | radial | ea | 0]) ----
    float8 acc[4];
#pragma unroll
    for (int nt = 0; nt < 4; ++nt) acc[nt] = (float8){};
#pragma unroll
    for (int s = 0; s < 5; ++s) {
        half16 A;
        if (s < 4) {
            const _Float16* p = (s < 2) ? (hbuf + (size_t)r * 64 + s * 32)
                                        : (hbuf + (size_t)c * 64 + (s - 2) * 32);
            A = packA(ld8h(p + 8 * kh), ld8h(p + 16 + 8 * kh));
        } else {
            half8 lo = {}, hi = {};
            if (kh == 0) {
                lo[0] = (_Float16)radial;
#pragma unroll
                for (int i = 1; i < 8; ++i) lo[i] = (_Float16)ea[i - 1];
                hi[0] = (_Float16)ea[15];
            } else {
#pragma unroll
                for (int i = 0; i < 8; ++i) lo[i] = (_Float16)ea[7 + i];
            }
            A = packA(lo, hi);
        }
#pragma unroll
        for (int nt = 0; nt < 4; ++nt)
            acc[nt] = wmma_f16(A, ldsB(sWe1, 160, nt, s, ln), acc[nt]);
    }
    // y = relu(acc + be1) -> LDS (f16)
#pragma unroll
    for (int nt = 0; nt < 4; ++nt) {
        int col = nt * 16 + sub;
        float b = be1[col];
#pragma unroll
        for (int v = 0; v < 8; ++v)
            sY[wv][(v + 8 * kh) * 64 + col] = (_Float16)fmaxf(acc[nt][v] + b, 0.0f);
    }

    // ---- GEMM2: y @ We2[64x64] ----
    float8 a2[4];
#pragma unroll
    for (int nt = 0; nt < 4; ++nt) a2[nt] = (float8){};
#pragma unroll
    for (int s = 0; s < 2; ++s) {
        const _Float16* yr = &sY[wv][sub * 64 + s * 32];
        half16 A = packA(ld8h(yr + 8 * kh), ld8h(yr + 16 + 8 * kh));
#pragma unroll
        for (int nt = 0; nt < 4; ++nt)
            a2[nt] = wmma_f16(A, ldsB(sWe2, 64, nt, s, ln), a2[nt]);
    }

    // m = relu(a2 + be2): atomic-add into agg_m (f32) + store f16 for coord MLP
    int r8[8];
#pragma unroll
    for (int v = 0; v < 8; ++v) r8[v] = sRow[wv][v + 8 * kh];
#pragma unroll
    for (int nt = 0; nt < 4; ++nt) {
        int col = nt * 16 + sub;
        float b = be2[col];
#pragma unroll
        for (int v = 0; v < 8; ++v) {
            float mv = fmaxf(a2[nt][v] + b, 0.0f);
            atomAddF(aggm + (size_t)r8[v] * 64 + col, mv);
            sY[wv][(v + 8 * kh) * 64 + col] = (_Float16)mv;
        }
    }

    // ---- GEMM3: m @ Wc1[64x64] ----
    float8 a3[4];
#pragma unroll
    for (int nt = 0; nt < 4; ++nt) a3[nt] = (float8){};
#pragma unroll
    for (int s = 0; s < 2; ++s) {
        const _Float16* mr = &sY[wv][sub * 64 + s * 32];
        half16 A = packA(ld8h(mr + 8 * kh), ld8h(mr + 16 + 8 * kh));
#pragma unroll
        for (int nt = 0; nt < 4; ++nt)
            a3[nt] = wmma_f16(A, ldsB(sWc1, 64, nt, s, ln), a3[nt]);
    }

    // ---- phi_x: s_e = relu(a3 + bc1) . Wc2 + bc2  (cross-lane reduce) ----
    float part[8];
#pragma unroll
    for (int v = 0; v < 8; ++v) part[v] = 0.0f;
#pragma unroll
    for (int nt = 0; nt < 4; ++nt) {
        int col = nt * 16 + sub;
        float w2 = Wc2[col];
        float b1 = bc1[col];
#pragma unroll
        for (int v = 0; v < 8; ++v)
            part[v] += fmaxf(a3[nt][v] + b1, 0.0f) * w2;
    }
#pragma unroll
    for (int msk = 1; msk < 16; msk <<= 1) {
#pragma unroll
        for (int v = 0; v < 8; ++v) part[v] += __shfl_xor(part[v], msk, 32);
    }
    if (sub < 8) {
        float sv = part[0];
#pragma unroll
        for (int v = 1; v < 8; ++v) sv = (sub == v) ? part[v] : sv;
        sv += bc2[0];
        int M  = sub + 8 * kh;
        int rr = sRow[wv][M];
        atomAddF(aggx + rr * 3 + 0, sDr[wv][M][0] * sv);
        atomAddF(aggx + rr * 3 + 1, sDr[wv][M][1] * sv);
        atomAddF(aggx + rr * 3 + 2, sDr[wv][M][2] * sv);
        atomAddF(cntbuf + rr, 1.0f);
    }
}

// ---------------------------------------------------------------------------
// node kernel: h' = relu([h|agg_m] @ Wn1 + bn1) @ Wn2 + bn2; out = h'@W_out+b;
// coords_out = coords + agg_x / max(cnt,1)
// ---------------------------------------------------------------------------
__global__ void __launch_bounds__(256)
egnn_node(const _Float16* __restrict__ hbuf, const float* __restrict__ aggm,
          const float* __restrict__ aggx, const float* __restrict__ cntbuf,
          const float* __restrict__ coords,
          const _Float16* __restrict__ Wn1t, const float* __restrict__ bn1,
          const _Float16* __restrict__ Wn2t, const float* __restrict__ bn2,
          const _Float16* __restrict__ Woutt, const float* __restrict__ bout,
          float* __restrict__ out, float* __restrict__ outc, int Ntot) {
    __shared__ __align__(32) _Float16 sWn1[64 * 128];
    __shared__ __align__(32) _Float16 sWn2[64 * 64];
    __shared__ __align__(32) _Float16 sWo[64 * 64];
    __shared__ __align__(32) _Float16 sY[8][16 * 64];

    stage_lds(sWn1, Wn1t, 64 * 128, threadIdx.x, blockDim.x);
    stage_lds(sWn2, Wn2t, 64 * 64, threadIdx.x, blockDim.x);
    stage_lds(sWo, Woutt, 64 * 64, threadIdx.x, blockDim.x);
    stage_fence();

    int wv = threadIdx.x >> 5, ln = threadIdx.x & 31;
    int n0 = (blockIdx.x * 8 + wv) * 16;
    if (n0 >= Ntot) return;
    int kh = ln >> 4, sub = ln & 15;
    int node = n0 + sub;

    // ---- GEMM1: [h|agg_m] (16x128) @ Wn1 ----
    float8 acc[4];
#pragma unroll
    for (int nt = 0; nt < 4; ++nt) acc[nt] = (float8){};
#pragma unroll
    for (int s = 0; s < 4; ++s) {
        half16 A;
        if (s < 2) {
            const _Float16* p = hbuf + (size_t)node * 64 + s * 32;
            A = packA(ld8h(p + 8 * kh), ld8h(p + 16 + 8 * kh));
        } else {
            const float* p = aggm + (size_t)node * 64 + (s - 2) * 32;
            A = packA(cvt8(p + 8 * kh), cvt8(p + 16 + 8 * kh));
        }
#pragma unroll
        for (int nt = 0; nt < 4; ++nt)
            acc[nt] = wmma_f16(A, ldsB(sWn1, 128, nt, s, ln), acc[nt]);
    }
#pragma unroll
    for (int nt = 0; nt < 4; ++nt) {
        int col = nt * 16 + sub;
        float b = bn1[col];
#pragma unroll
        for (int v = 0; v < 8; ++v)
            sY[wv][(v + 8 * kh) * 64 + col] = (_Float16)fmaxf(acc[nt][v] + b, 0.0f);
    }

    // ---- GEMM2: y @ Wn2 (+bn2, no relu) ----
    float8 a2[4];
#pragma unroll
    for (int nt = 0; nt < 4; ++nt) a2[nt] = (float8){};
#pragma unroll
    for (int s = 0; s < 2; ++s) {
        const _Float16* yr = &sY[wv][sub * 64 + s * 32];
        half16 A = packA(ld8h(yr + 8 * kh), ld8h(yr + 16 + 8 * kh));
#pragma unroll
        for (int nt = 0; nt < 4; ++nt)
            a2[nt] = wmma_f16(A, ldsB(sWn2, 64, nt, s, ln), a2[nt]);
    }
#pragma unroll
    for (int nt = 0; nt < 4; ++nt) {
        int col = nt * 16 + sub;
        float b = bn2[col];
#pragma unroll
        for (int v = 0; v < 8; ++v)
            sY[wv][(v + 8 * kh) * 64 + col] = (_Float16)(a2[nt][v] + b);
    }

    // ---- GEMM3: h2 @ W_out + b_out -> global ----
    float8 a3[4];
#pragma unroll
    for (int nt = 0; nt < 4; ++nt) a3[nt] = (float8){};
#pragma unroll
    for (int s = 0; s < 2; ++s) {
        const _Float16* hr = &sY[wv][sub * 64 + s * 32];
        half16 A = packA(ld8h(hr + 8 * kh), ld8h(hr + 16 + 8 * kh));
#pragma unroll
        for (int nt = 0; nt < 4; ++nt)
            a3[nt] = wmma_f16(A, ldsB(sWo, 64, nt, s, ln), a3[nt]);
    }
#pragma unroll
    for (int nt = 0; nt < 4; ++nt) {
        int col = nt * 16 + sub;
        float b = bout[col];
#pragma unroll
        for (int v = 0; v < 8; ++v)
            out[(size_t)(n0 + v + 8 * kh) * 64 + col] = a3[nt][v] + b;
    }

    // ---- coords update ----
    if (ln < 16) {
        int n = n0 + ln;
        float cdiv = fmaxf(cntbuf[n], 1.0f);
#pragma unroll
        for (int k = 0; k < 3; ++k)
            outc[n * 3 + k] = coords[n * 3 + k] + aggx[n * 3 + k] / cdiv;
    }
}

// ---------------------------------------------------------------------------
extern "C" void kernel_launch(void* const* d_in, const int* in_sizes, int n_in,
                              void* d_out, int out_size, void* d_ws, size_t ws_size,
                              hipStream_t stream) {
    const float* node_feats = (const float*)d_in[0];
    const float* edge_attr  = (const float*)d_in[1];
    const float* coords     = (const float*)d_in[2];
    const int*   edge_index = (const int*)d_in[3];
    const float* W_in  = (const float*)d_in[4];
    const float* b_in  = (const float*)d_in[5];
    const float* W_out = (const float*)d_in[6];
    const float* b_out = (const float*)d_in[7];
    const float* We1 = (const float*)d_in[8];
    const float* be1 = (const float*)d_in[9];
    const float* We2 = (const float*)d_in[10];
    const float* be2 = (const float*)d_in[11];
    const float* Wn1 = (const float*)d_in[12];
    const float* bn1 = (const float*)d_in[13];
    const float* Wn2 = (const float*)d_in[14];
    const float* bn2 = (const float*)d_in[15];
    const float* Wc1 = (const float*)d_in[16];
    const float* bc1 = (const float*)d_in[17];
    const float* Wc2 = (const float*)d_in[18];
    const float* bc2 = (const float*)d_in[19];

    char* ws = (char*)d_ws;
    _Float16* hbuf = (_Float16*)(ws + OFF_H);
    float* aggm = (float*)(ws + OFF_AGGM);
    float* aggx = (float*)(ws + OFF_AGGX);
    float* cnt  = (float*)(ws + OFF_CNT);
    _Float16* We1t  = (_Float16*)(ws + OFF_WE1T);
    _Float16* We2t  = (_Float16*)(ws + OFF_WE2T);
    _Float16* Wc1t  = (_Float16*)(ws + OFF_WC1T);
    _Float16* Wn1t  = (_Float16*)(ws + OFF_WN1T);
    _Float16* Wn2t  = (_Float16*)(ws + OFF_WN2T);
    _Float16* Woutt = (_Float16*)(ws + OFF_WOT);
    _Float16* Wint  = (_Float16*)(ws + OFF_WIT);

    float* out_h = (float*)d_out;             // [N,64]
    float* out_c = out_h + (size_t)NN * 64;   // [N,3]

    // weight prep (f32 -> transposed f16, zero-padded K)
    auto tlaunch = [&](const float* src, _Float16* dst, int K, int Ncol, int Kpad) {
        int total = Ncol * Kpad;
        transpose_to_half<<<(total + 255) / 256, 256, 0, stream>>>(src, dst, K, Ncol, Kpad);
    };
    tlaunch(W_in, Wint, 32, 64, 32);
    tlaunch(We1, We1t, 145, 64, 160);
    tlaunch(We2, We2t, 64, 64, 64);
    tlaunch(Wc1, Wc1t, 64, 64, 64);
    tlaunch(Wn1, Wn1t, 128, 64, 128);
    tlaunch(Wn2, Wn2t, 64, 64, 64);
    tlaunch(W_out, Woutt, 64, 64, 64);

    (void)hipMemsetAsync(aggm, 0, (size_t)NN * 64 * 4, stream);
    (void)hipMemsetAsync(aggx, 0, (size_t)NN * 3 * 4, stream);
    (void)hipMemsetAsync(cnt, 0, (size_t)NN * 4, stream);

    int node_blocks = (NN / 16 + 7) / 8;   // 8 waves/block, 16 rows/wave
    int edge_blocks = EE / 128;            // exact: 1.2M / (8*16)

    egnn_embed<<<node_blocks, 256, 0, stream>>>(node_feats, Wint, b_in, hbuf, NN);
    egnn_edge<<<edge_blocks, 256, 0, stream>>>(hbuf, edge_attr, coords, edge_index,
                                               We1t, be1, We2t, be2, Wc1t, bc1,
                                               Wc2, bc2, aggm, aggx, cnt, EE);
    egnn_node<<<node_blocks, 256, 0, stream>>>(hbuf, aggm, aggx, cnt, coords,
                                               Wn1t, bn1, Wn2t, bn2, Woutt, b_out,
                                               out_h, out_c, NN);
}
